// SGNN_69492570849816
// MI455X (gfx1250) — matrix operand order repaired and checked by reading, not compiled, not run on hardware
//
#include <hip/hip_runtime.h>

#define H       256
#define KNB     30
#define LSEQ    1024
#define BATCH   4
#define NNODES  (BATCH*LSEQ)      // 4096
#define NEDGES  (NNODES*KNB)      // 122880
#define NLAYERS 4

typedef __attribute__((ext_vector_type(16))) __bf16 v16bf;
typedef __attribute__((ext_vector_type(8)))  __bf16 bf16x8;
typedef __attribute__((ext_vector_type(8)))  float  v8f;

// ---------- scalar bf16 helpers (bit-level, RNE) ----------
__device__ inline __bf16 f2bf(float f){
  unsigned u; __builtin_memcpy(&u,&f,4);
  u += 0x7FFFu + ((u>>16)&1u);
  unsigned short h=(unsigned short)(u>>16);
  __bf16 r; __builtin_memcpy(&r,&h,2); return r;
}
__device__ inline float bf2f(__bf16 b){
  unsigned short h; __builtin_memcpy(&h,&b,2);
  unsigned u=((unsigned)h)<<16;
  float f; __builtin_memcpy(&f,&u,4); return f;
}

__device__ inline v8f wmma_bf16(v16bf a, v16bf b, v8f c){
  return __builtin_amdgcn_wmma_f32_16x16x32_bf16(false, a, false, b, (short)0, c, false, false);
}

// A fragment (16x32) from LDS staged in k16-swizzled layout:
// element (row,k) at [((k>>4)*rows + row)*16 + (k&15)]
__device__ inline v16bf load_a(const __bf16* base, int rows, int ks, int row, int hgrp){
  bf16x8 lo = *(const bf16x8*)(base + (((2*ks + hgrp)*rows + row)<<4));
  bf16x8 hi = *(const bf16x8*)(base + (((2*ks + hgrp)*rows + row)<<4) + 8);
  v16bf a;
#pragma unroll
  for (int i=0;i<8;++i){ a[i]=lo[i]; a[i+8]=hi[i]; }
  return a;
}
// B fragment (32x16) from pre-swizzled global weights: one 32B contiguous load per lane
__device__ inline v16bf load_b(const __bf16* w, int N, int ks, int col, int sel){
  return *(const v16bf*)(w + ((size_t)((2*ks + sel)*N + col)<<4));
}
__device__ inline void store_act(__bf16* dst, int rows, int row, int col, float v){
  dst[(((col>>4)*rows + row)<<4) + (col&15)] = f2bf(v);
}

__device__ inline float block_sum256(float v, float* red){
  __syncthreads();
#pragma unroll
  for (int off=16; off>0; off>>=1) v += __shfl_down(v, off, 32);
  const int lane = threadIdx.x & 31, w = threadIdx.x >> 5;
  if (lane==0) red[w] = v;
  __syncthreads();
  if (w==0){
    float r = (lane<8) ? red[lane] : 0.f;
#pragma unroll
    for (int off=4; off>0; off>>=1) r += __shfl_down(r, off, 32);
    if (lane==0) red[0] = r;
  }
  __syncthreads();
  return red[0];
}

// ---------- weight pre-swizzle fp32 -> bf16 B-fragment layout ----------
__global__ __launch_bounds__(256) void swz_kernel(const float* __restrict__ src,
                                                  __bf16* __restrict__ dst, int K, int N){
  int i = blockIdx.x*256 + threadIdx.x;
  if (i >= K*N) return;
  int k = i / N, n = i - k*N;
  dst[(((k>>4)*N + n)<<4) + (k&15)] = f2bf(src[i]);
}

// ---------- input projections ----------
__global__ __launch_bounds__(256) void projV_kernel(const float* __restrict__ nodes,
    const float* __restrict__ src, const int* __restrict__ mask,
    const float* __restrict__ Wv, const float* __restrict__ bv,
    float* __restrict__ hV){
  __shared__ float x[40];
  const int n = blockIdx.x, tid = threadIdx.x;
  if (tid < 10) x[tid] = nodes[n*10 + tid];
  else if (tid < 40) x[tid] = (mask[n]==1) ? src[n*30 + (tid-10)] : 0.f;
  __syncthreads();
  float s = bv[tid];
#pragma unroll
  for (int i=0;i<40;++i) s += x[i]*Wv[i*H + tid];
  hV[n*H + tid] = s;
}

__global__ __launch_bounds__(256) void projE_kernel(const float* __restrict__ edges,
    const float* __restrict__ We, const float* __restrict__ be,
    __bf16* __restrict__ hE){
  __shared__ float x[11];
  const long long e = blockIdx.x;
  const int tid = threadIdx.x;
  if (tid < 11) x[tid] = edges[e*11 + tid];
  __syncthreads();
  float s = be[tid];
#pragma unroll
  for (int i=0;i<11;++i) s += x[i]*We[i*H + tid];
  hE[e*H + tid] = f2bf(s);
}

// ---------- fused message MLP + mean + residual + LN : one block per node ----------
__global__ __launch_bounds__(256) void msg_kernel(const float* __restrict__ hVin,
    const __bf16* __restrict__ hEbf, const int* __restrict__ conn,
    const __bf16* __restrict__ W1, const float* __restrict__ b1,
    const __bf16* __restrict__ W2, const float* __restrict__ b2,
    const __bf16* __restrict__ W3, const float* __restrict__ b3,
    const float* __restrict__ g1, const float* __restrict__ be1,
    float* __restrict__ hVout){
  __shared__ __align__(32) char lds[65536];
  __bf16* A_sw = (__bf16*)lds;               // [0,48K)  staged hEV (32x768)
  __bf16* act2 = (__bf16*)lds;               // [0,16K)  aliases A after GEMM1
  float*  S    = (float*) (lds + 16384);     // [16K,48K) GEMM3 out 32x256 f32
  __bf16* act1 = (__bf16*)(lds + 49152);     // [48K,64K)
  int*    cnS  = (int*)   (lds + 49152);     // staging-time alias
  float*  red  = (float*) (lds + 49152);     // LN-time alias

  const int tid = threadIdx.x;
  const int n   = blockIdx.x;
  const int bix = n >> 10;                   // batch index (L = 1024)

  if (tid < KNB) cnS[tid] = bix*LSEQ + conn[n*KNB + tid];
  __syncthreads();

  // stage A = [hV(self) | hE | hV(neighbor)] as bf16, swizzled, rows>=30 zero
  {
    const int row   = tid >> 3;          // 0..31
    const int cbase = (tid & 7) * 96;    // 96 cols/thread
#pragma unroll 4
    for (int j = 0; j < 96; ++j) {
      int c = cbase + j;
      float v = 0.f;
      if (row < KNB) {
        if (c < 256)      v = hVin[n*H + c];
        else if (c < 512) v = bf2f(hEbf[(n*KNB + row)*H + (c-256)]);
        else              v = hVin[cnS[row]*H + (c-512)];
      }
      A_sw[(((c>>4)*32 + row)<<4) + (c&15)] = f2bf(v);
    }
  }
  __syncthreads();

  const int lane = tid & 31, wv = tid >> 5;
  const int hgrp = lane >> 4, mrow = lane & 15;
  const int colw = wv * 32;
  const int col0 = colw + mrow, col1 = colw + 16 + mrow;
  const v8f vz = {};
  v8f a00, a01, a10, a11;

  // GEMM1: 32x768 @ 768x256, relu
  a00=vz; a01=vz; a10=vz; a11=vz;
  for (int ks = 0; ks < 24; ++ks) {
    v16bf fa0 = load_a(A_sw, 32, ks, mrow,      hgrp);
    v16bf fa1 = load_a(A_sw, 32, ks, mrow + 16, hgrp);
    v16bf fb0 = load_b(W1, 256, ks, col0, hgrp);
    v16bf fb1 = load_b(W1, 256, ks, col1, hgrp);
    a00 = wmma_bf16(fa0, fb0, a00);  a01 = wmma_bf16(fa0, fb1, a01);
    a10 = wmma_bf16(fa1, fb0, a10);  a11 = wmma_bf16(fa1, fb1, a11);
  }
  {
    float bb0 = b1[col0], bb1 = b1[col1];
#pragma unroll
    for (int g=0; g<8; ++g){
      int r = g + 8*hgrp;
      store_act(act1, 32, r,    col0, fmaxf(a00[g]+bb0, 0.f));
      store_act(act1, 32, r,    col1, fmaxf(a01[g]+bb1, 0.f));
      store_act(act1, 32, r+16, col0, fmaxf(a10[g]+bb0, 0.f));
      store_act(act1, 32, r+16, col1, fmaxf(a11[g]+bb1, 0.f));
    }
  }
  __syncthreads();

  // GEMM2: 32x256 @ 256x256, relu
  a00=vz; a01=vz; a10=vz; a11=vz;
  for (int ks = 0; ks < 8; ++ks) {
    v16bf fa0 = load_a(act1, 32, ks, mrow,      hgrp);
    v16bf fa1 = load_a(act1, 32, ks, mrow + 16, hgrp);
    v16bf fb0 = load_b(W2, 256, ks, col0, hgrp);
    v16bf fb1 = load_b(W2, 256, ks, col1, hgrp);
    a00 = wmma_bf16(fa0, fb0, a00);  a01 = wmma_bf16(fa0, fb1, a01);
    a10 = wmma_bf16(fa1, fb0, a10);  a11 = wmma_bf16(fa1, fb1, a11);
  }
  {
    float bb0 = b2[col0], bb1 = b2[col1];
#pragma unroll
    for (int g=0; g<8; ++g){
      int r = g + 8*hgrp;
      store_act(act2, 32, r,    col0, fmaxf(a00[g]+bb0, 0.f));
      store_act(act2, 32, r,    col1, fmaxf(a01[g]+bb1, 0.f));
      store_act(act2, 32, r+16, col0, fmaxf(a10[g]+bb0, 0.f));
      store_act(act2, 32, r+16, col1, fmaxf(a11[g]+bb1, 0.f));
    }
  }
  __syncthreads();

  // GEMM3: 32x256 @ 256x256, +bias -> S (f32)
  a00=vz; a01=vz; a10=vz; a11=vz;
  for (int ks = 0; ks < 8; ++ks) {
    v16bf fa0 = load_a(act2, 32, ks, mrow,      hgrp);
    v16bf fa1 = load_a(act2, 32, ks, mrow + 16, hgrp);
    v16bf fb0 = load_b(W3, 256, ks, col0, hgrp);
    v16bf fb1 = load_b(W3, 256, ks, col1, hgrp);
    a00 = wmma_bf16(fa0, fb0, a00);  a01 = wmma_bf16(fa0, fb1, a01);
    a10 = wmma_bf16(fa1, fb0, a10);  a11 = wmma_bf16(fa1, fb1, a11);
  }
  {
    float bb0 = b3[col0], bb1 = b3[col1];
#pragma unroll
    for (int g=0; g<8; ++g){
      int r = g + 8*hgrp;
      S[r*256 + col0]      = a00[g] + bb0;
      S[r*256 + col1]      = a01[g] + bb1;
      S[(r+16)*256 + col0] = a10[g] + bb0;
      S[(r+16)*256 + col1] = a11[g] + bb1;
    }
  }
  __syncthreads();

  // masked mean over 30 real edges + residual + LayerNorm (channel = tid)
  float x;
  {
    float s = 0.f;
#pragma unroll
    for (int k=0;k<KNB;++k) s += S[k*256 + tid];
    x = hVin[n*H + tid] + s * (1.0f/KNB);
  }
  float mean = block_sum256(x, red) * (1.f/H);
  float d = x - mean;
  float var  = block_sum256(d*d, red) * (1.f/H);
  hVout[n*H + tid] = d * rsqrtf(var + 1e-5f) * g1[tid] + be1[tid];
}

// ---------- fused FFN + residual + LN : 16 nodes per block ----------
__global__ __launch_bounds__(256) void ffn_kernel(const float* __restrict__ hVin,
    const __bf16* __restrict__ Win,  const float* __restrict__ bin,
    const __bf16* __restrict__ Wout, const float* __restrict__ bout,
    const float* __restrict__ g2, const float* __restrict__ be2,
    float* __restrict__ hVout){
  __shared__ __align__(32) char lds[8192 + 32768 + 16384 + 64];
  __bf16* A_sw = (__bf16*)lds;                      // 16 x 256
  __bf16* act  = (__bf16*)(lds + 8192);             // 16 x 1024
  float*  S    = (float*) (lds + 8192 + 32768);     // 16 x 256
  float*  red  = (float*) (lds + 8192 + 32768 + 16384);

  const int tid = threadIdx.x;
  const int n0  = blockIdx.x * 16;

  for (int idx = tid; idx < 16*256; idx += 256) {
    int row = idx >> 8, c = idx & 255;
    store_act(A_sw, 16, row, c, hVin[(n0+row)*H + c]);
  }
  __syncthreads();

  const int lane = tid & 31, wv = tid >> 5;
  const int hgrp = lane >> 4, mrow = lane & 15;
  const v8f vz = {};

  // GEMM-A: 16x256 @ 256x1024, relu; wave cols [128*wv, +128)
  v8f acc[8];
#pragma unroll
  for (int t=0;t<8;++t) acc[t] = vz;
  for (int ks = 0; ks < 8; ++ks) {
    v16bf fa = load_a(A_sw, 16, ks, mrow, hgrp);
#pragma unroll
    for (int t=0;t<8;++t) {
      v16bf fb = load_b(Win, 1024, ks, 128*wv + 16*t + mrow, hgrp);
      acc[t] = wmma_bf16(fa, fb, acc[t]);
    }
  }
#pragma unroll
  for (int t=0;t<8;++t){
    int col = 128*wv + 16*t + mrow;
    float bv = bin[col];
#pragma unroll
    for (int g=0; g<8; ++g) store_act(act, 16, g + 8*hgrp, col, fmaxf(acc[t][g] + bv, 0.f));
  }
  __syncthreads();

  // GEMM-B: 16x1024 @ 1024x256; wave cols [32*wv, +32)
  v8f c0 = vz, c1 = vz;
  for (int ks=0; ks<32; ++ks){
    v16bf fa = load_a(act, 16, ks, mrow, hgrp);
    v16bf fb0 = load_b(Wout, 256, ks, 32*wv + mrow, hgrp);
    v16bf fb1 = load_b(Wout, 256, ks, 32*wv + 16 + mrow, hgrp);
    c0 = wmma_bf16(fa, fb0, c0);
    c1 = wmma_bf16(fa, fb1, c1);
  }
  {
    int col0 = 32*wv + mrow, col1 = col0 + 16;
    float b0 = bout[col0], b1v = bout[col1];
#pragma unroll
    for (int g=0; g<8; ++g){
      int r = g + 8*hgrp;
      S[r*256 + col0] = c0[g] + b0;
      S[r*256 + col1] = c1[g] + b1v;
    }
  }
  __syncthreads();

  for (int i=0;i<16;++i){
    float x = S[i*256 + tid] + hVin[(n0+i)*H + tid];
    float mean = block_sum256(x, red) * (1.f/H);
    float d = x - mean;
    float var  = block_sum256(d*d, red) * (1.f/H);
    hVout[(n0+i)*H + tid] = d * rsqrtf(var + 1e-5f) * g2[tid] + be2[tid];
  }
}

// ---------- pooling + dense head (tiny, fp32) ----------
__global__ __launch_bounds__(256) void head_kernel(const float* __restrict__ hV,
    const int* __restrict__ mask, const int* __restrict__ lengths,
    const float* __restrict__ D1w, const float* __restrict__ D1b,
    const float* __restrict__ lng, const float* __restrict__ lnb,
    const float* __restrict__ D2w, const float* __restrict__ D2b,
    float* __restrict__ out){
  __shared__ float p[256];
  __shared__ float h1[1024];
  __shared__ float red[8];
  const int b = blockIdx.x, tid = threadIdx.x;
  float s = 0.f;
  for (int l=0;l<LSEQ;++l){
    int idx = b*LSEQ + l;
    if (mask[idx] == 1) s += hV[idx*H + tid];
  }
  p[tid] = s / (float)lengths[b];
  __syncthreads();
  for (int j=tid;j<1024;j+=256){
    float a = D1b[j];
    for (int i=0;i<256;++i) a += p[i]*D1w[i*1024 + j];
    h1[j] = fmaxf(a, 0.f);
  }
  float lsum = 0.f;
  for (int j=tid;j<1024;j+=256) lsum += h1[j];
  float mean = block_sum256(lsum, red) * (1.f/1024);
  float vsum = 0.f;
  for (int j=tid;j<1024;j+=256){ float d = h1[j]-mean; vsum += d*d; }
  float var = block_sum256(vsum, red) * (1.f/1024);
  float r = rsqrtf(var + 1e-5f);
  float o = 0.f;
  for (int j=tid;j<1024;j+=256){
    float hn = (h1[j]-mean)*r*lng[j] + lnb[j];
    o += hn * D2w[j];
  }
  o = block_sum256(o, red);
  if (tid==0) out[b] = o + D2b[0] + 0.5f;
}

extern "C" void kernel_launch(void* const* d_in, const int* in_sizes, int n_in,
                              void* d_out, int out_size, void* d_ws, size_t ws_size,
                              hipStream_t stream) {
  const float* nodes   = (const float*)d_in[0];
  const float* edges   = (const float*)d_in[1];
  const int*   conn    = (const int*)  d_in[2];
  const float* src     = (const float*)d_in[3];
  const int*   mask    = (const int*)  d_in[4];
  const int*   lengths = (const int*)  d_in[5];
  const float* Wv_w = (const float*)d_in[6];
  const float* Wv_b = (const float*)d_in[7];
  const float* We_w = (const float*)d_in[8];
  const float* We_b = (const float*)d_in[9];
  const float* W1w  = (const float*)d_in[10];
  const float* W1b  = (const float*)d_in[11];
  const float* W2w  = (const float*)d_in[12];
  const float* W2b  = (const float*)d_in[13];
  const float* W3w  = (const float*)d_in[14];
  const float* W3b  = (const float*)d_in[15];
  const float* n1g  = (const float*)d_in[16];
  const float* n1b  = (const float*)d_in[17];
  const float* Winw = (const float*)d_in[18];
  const float* Winb = (const float*)d_in[19];
  const float* Wow  = (const float*)d_in[20];
  const float* Wob  = (const float*)d_in[21];
  const float* n2g  = (const float*)d_in[22];
  const float* n2b  = (const float*)d_in[23];
  const float* D1w  = (const float*)d_in[24];
  const float* D1b  = (const float*)d_in[25];
  const float* lng  = (const float*)d_in[26];
  const float* lnb  = (const float*)d_in[27];
  const float* D2w  = (const float*)d_in[28];
  const float* D2b  = (const float*)d_in[29];

  // workspace layout (bytes)
  char* ws = (char*)d_ws;
  const size_t off_hE   = 0;                                  // 122880*256*2 = 62,914,560
  const size_t off_W1   = off_hE   + (size_t)NEDGES*H*2;      // 4*768*256*2
  const size_t off_W2   = off_W1   + (size_t)NLAYERS*768*256*2;
  const size_t off_W3   = off_W2   + (size_t)NLAYERS*256*256*2;
  const size_t off_Win  = off_W3   + (size_t)NLAYERS*256*256*2;
  const size_t off_Wout = off_Win  + (size_t)NLAYERS*256*1024*2;
  const size_t off_hVa  = off_Wout + (size_t)NLAYERS*1024*256*2;
  const size_t off_hVb  = off_hVa  + (size_t)NNODES*H*4;
  __bf16* hEbf  = (__bf16*)(ws + off_hE);
  __bf16* W1s   = (__bf16*)(ws + off_W1);
  __bf16* W2s   = (__bf16*)(ws + off_W2);
  __bf16* W3s   = (__bf16*)(ws + off_W3);
  __bf16* Wins  = (__bf16*)(ws + off_Win);
  __bf16* Wouts = (__bf16*)(ws + off_Wout);
  float*  hVa   = (float*) (ws + off_hVa);
  float*  hVb   = (float*) (ws + off_hVb);

  // pre-swizzle weights to bf16 B-fragment layout
  for (int l = 0; l < NLAYERS; ++l) {
    swz_kernel<<<(768*256+255)/256,  256, 0, stream>>>(W1w  + (size_t)l*768*256,  W1s   + (size_t)l*768*256,  768,  256);
    swz_kernel<<<(256*256+255)/256,  256, 0, stream>>>(W2w  + (size_t)l*256*256,  W2s   + (size_t)l*256*256,  256,  256);
    swz_kernel<<<(256*256+255)/256,  256, 0, stream>>>(W3w  + (size_t)l*256*256,  W3s   + (size_t)l*256*256,  256,  256);
    swz_kernel<<<(256*1024+255)/256, 256, 0, stream>>>(Winw + (size_t)l*256*1024, Wins  + (size_t)l*256*1024, 256, 1024);
    swz_kernel<<<(1024*256+255)/256, 256, 0, stream>>>(Wow  + (size_t)l*1024*256, Wouts + (size_t)l*1024*256, 1024, 256);
  }

  projV_kernel<<<NNODES, 256, 0, stream>>>(nodes, src, mask, Wv_w, Wv_b, hVa);
  projE_kernel<<<NEDGES, 256, 0, stream>>>(edges, We_w, We_b, hEbf);

  for (int l = 0; l < NLAYERS; ++l) {
    msg_kernel<<<NNODES, 256, 0, stream>>>(hVa, hEbf, conn,
        W1s + (size_t)l*768*256, W1b + l*256,
        W2s + (size_t)l*256*256, W2b + l*256,
        W3s + (size_t)l*256*256, W3b + l*256,
        n1g + l*256, n1b + l*256, hVb);
    ffn_kernel<<<NNODES/16, 256, 0, stream>>>(hVb,
        Wins  + (size_t)l*256*1024, Winb + l*1024,
        Wouts + (size_t)l*1024*256, Wob  + l*256,
        n2g + l*256, n2b + l*256, hVa);
  }

  head_kernel<<<BATCH, 256, 0, stream>>>(hVa, mask, lengths,
      D1w, D1b, lng, lnb, D2w, D2b, (float*)d_out);
  (void)in_sizes; (void)n_in; (void)out_size; (void)ws_size;
}
